// S6_noC_53730040873390
// MI455X (gfx1250) — matrix operand order repaired and checked by reading, not compiled, not run on hardware
//
#include <hip/hip_runtime.h>

#define BATCH  8
#define SEQL   2048
#define DMODEL 1024
#define MROWS  (BATCH * SEQL)      // 16384
#define NCH    16                  // scan chunks per sequence
#define CHLEN  (SEQL / NCH)        // 128

typedef __attribute__((ext_vector_type(16))) __bf16 bf16x16;
typedef __attribute__((ext_vector_type(8)))  __bf16 bf16x8;
typedef __attribute__((ext_vector_type(8)))  float  v8f;

__device__ __forceinline__ float sigmoidf_(float x) {
    return 1.0f / (1.0f + __expf(-x));
}

// ---------------- fp32 -> bf16 convert (grid-stride) ----------------
__global__ void k_f32_to_bf16(const float* __restrict__ src,
                              __bf16* __restrict__ dst, int n) {
    int i = blockIdx.x * blockDim.x + threadIdx.x;
    int stride = gridDim.x * blockDim.x;
    for (; i < n; i += stride) dst[i] = (__bf16)src[i];
}

// ---------------- mean over sequence: sq[b,d] = mean_l x[b,l,d] ----------------
__global__ void k_mean_seq(const float* __restrict__ x, float* __restrict__ sq) {
    int tid = blockIdx.x * blockDim.x + threadIdx.x;   // 8192 = B*D
    int b = tid >> 10, d = tid & (DMODEL - 1);
    const float* p = x + (size_t)b * SEQL * DMODEL + d;
    float s = 0.f;
    for (int l = 0; l < SEQL; ++l) s += p[(size_t)l * DMODEL];
    sq[tid] = s * (1.0f / SEQL);
}

// ---------------- SToken: tf[b,d] = sigmoid(relu(sq@Wst^T + bst)) + st_bias ----------------
__global__ void k_stoken(const float* __restrict__ sq, const float* __restrict__ Wst,
                         const float* __restrict__ bst, const float* __restrict__ stbias,
                         float* __restrict__ tf) {
    int tid = blockIdx.x * blockDim.x + threadIdx.x;   // 8192 = B*D
    int b = tid >> 10, n = tid & (DMODEL - 1);
    const float* q = sq + (size_t)b * DMODEL;
    const float* w = Wst + (size_t)n * DMODEL;
    float acc = 0.f;
    for (int k = 0; k < DMODEL; ++k) acc = fmaf(q[k], w[k], acc);
    float v = acc + bst[n];
    v = v > 0.f ? v : 0.f;
    tf[tid] = sigmoidf_(v) + stbias[n];
}

// ---------------- fused triple GEMM epilogue ----------------
__device__ __forceinline__ void epi3(int mt, int half, int col,
                                     v8f aB, v8f aC, v8f aD,
                                     float biasB, float biasC, float biasD,
                                     const float* __restrict__ dp,
                                     float* __restrict__ oB0, float* __restrict__ oC,
                                     float* __restrict__ oDl) {
#pragma unroll
    for (int v = 0; v < 8; ++v) {
        int row = mt * 16 + half * 8 + v;          // global M row (b*L + l)
        int l = row & (SEQL - 1);
        size_t g = (size_t)row * DMODEL + col;
        oB0[g] = aB[v] + biasB;
        oC[g]  = aC[v] + biasC;
        oDl[g] = sigmoidf_(aD[v] + biasD + dp[(size_t)l * DMODEL + col]);
    }
}

#define WMMA_BF16(a, b, c) \
    __builtin_amdgcn_wmma_f32_16x16x32_bf16(false, (a), false, (b), (short)0, (c), false, false)

// ---------------- B0 = x@W_B^T, C = x@W_C^T, delta = sigmoid(x@W_d^T + b_d + dp) ----------------
// Block = 256 threads = 8 waves; each wave: 2 M-tiles x 2 N-tiles x 3 weight matrices
// (12 accumulators, 12 WMMAs per K-step against 8 b256 fragment loads).
__global__ __launch_bounds__(256)
void k_gemm3_wmma(const __bf16* __restrict__ xb,
                  const __bf16* __restrict__ wB, const __bf16* __restrict__ wC,
                  const __bf16* __restrict__ wD,
                  const float* __restrict__ bB, const float* __restrict__ bC,
                  const float* __restrict__ bD, const float* __restrict__ delta_p,
                  float* __restrict__ outB0, float* __restrict__ outC,
                  float* __restrict__ outDelta) {
    const int K = DMODEL;
    int nt = blockIdx.x & 31;           // 32 N-groups of 32 columns
    int mblock = blockIdx.x >> 5;       // 64 M-blocks of 256 rows
    int wave = threadIdx.x >> 5;
    int lane = threadIdx.x & 31;
    int half = lane >> 4;               // which 16-lane half of the wave
    int nl   = lane & 15;

    int col0 = nt * 32 + nl;            // first N-tile column (also W row)
    int col1 = col0 + 16;               // second N-tile column
    int mt0  = mblock * 16 + wave * 2;  // first of this wave's two 16-row M-tiles
    int row0 = mt0 * 16 + nl;
    int row1 = row0 + 16;

    const __bf16* xr0 = xb + (size_t)row0 * K;
    const __bf16* xr1 = xb + (size_t)row1 * K;
    const __bf16* wrB0 = wB + (size_t)col0 * K + half * 16;
    const __bf16* wrC0 = wC + (size_t)col0 * K + half * 16;
    const __bf16* wrD0 = wD + (size_t)col0 * K + half * 16;
    const __bf16* wrB1 = wB + (size_t)col1 * K + half * 16;
    const __bf16* wrC1 = wC + (size_t)col1 * K + half * 16;
    const __bf16* wrD1 = wD + (size_t)col1 * K + half * 16;

    v8f aB00 = {}, aC00 = {}, aD00 = {};   // M-tile0 x N-tile0
    v8f aB01 = {}, aC01 = {}, aD01 = {};   // M-tile0 x N-tile1
    v8f aB10 = {}, aC10 = {}, aD10 = {};   // M-tile1 x N-tile0
    v8f aB11 = {}, aC11 = {}, aD11 = {};   // M-tile1 x N-tile1

    for (int kk = 0; kk < K; kk += 32) {
        __builtin_prefetch(xr0 + kk + 256, 0, 1);
        __builtin_prefetch(xr1 + kk + 256, 0, 1);
        // B fragments: 16 contiguous bf16 of W rows col0/col1, K-offset kk + half*16
        bf16x16 fB0 = *(const bf16x16*)(wrB0 + kk);
        bf16x16 fC0 = *(const bf16x16*)(wrC0 + kk);
        bf16x16 fD0 = *(const bf16x16*)(wrD0 + kk);
        bf16x16 fB1 = *(const bf16x16*)(wrB1 + kk);
        bf16x16 fC1 = *(const bf16x16*)(wrC1 + kk);
        bf16x16 fD1 = *(const bf16x16*)(wrD1 + kk);
        // A fragments: 16 contiguous bf16 of rows row0/row1, K-offset kk + half*16 (two 8-element chunks)
        bf16x8 l0 = *(const bf16x8*)(xr0 + kk + half * 16);
        bf16x8 h0 = *(const bf16x8*)(xr0 + kk + half * 16 + 8);
        bf16x8 l1 = *(const bf16x8*)(xr1 + kk + half * 16);
        bf16x8 h1 = *(const bf16x8*)(xr1 + kk + half * 16 + 8);
        bf16x16 a0 = __builtin_shufflevector(l0, h0, 0,1,2,3,4,5,6,7,8,9,10,11,12,13,14,15);
        bf16x16 a1 = __builtin_shufflevector(l1, h1, 0,1,2,3,4,5,6,7,8,9,10,11,12,13,14,15);

        aB00 = WMMA_BF16(a0, fB0, aB00);
        aC00 = WMMA_BF16(a0, fC0, aC00);
        aD00 = WMMA_BF16(a0, fD0, aD00);
        aB01 = WMMA_BF16(a0, fB1, aB01);
        aC01 = WMMA_BF16(a0, fC1, aC01);
        aD01 = WMMA_BF16(a0, fD1, aD01);
        aB10 = WMMA_BF16(a1, fB0, aB10);
        aC10 = WMMA_BF16(a1, fC0, aC10);
        aD10 = WMMA_BF16(a1, fD0, aD10);
        aB11 = WMMA_BF16(a1, fB1, aB11);
        aC11 = WMMA_BF16(a1, fC1, aC11);
        aD11 = WMMA_BF16(a1, fD1, aD11);
    }

    float biasB0 = bB[col0], biasC0 = bC[col0], biasD0 = bD[col0];
    float biasB1 = bB[col1], biasC1 = bC[col1], biasD1 = bD[col1];
    epi3(mt0,     half, col0, aB00, aC00, aD00, biasB0, biasC0, biasD0, delta_p, outB0, outC, outDelta);
    epi3(mt0,     half, col1, aB01, aC01, aD01, biasB1, biasC1, biasD1, delta_p, outB0, outC, outDelta);
    epi3(mt0 + 1, half, col0, aB10, aC10, aD10, biasB0, biasC0, biasD0, delta_p, outB0, outC, outDelta);
    epi3(mt0 + 1, half, col1, aB11, aC11, aD11, biasB1, biasC1, biasD1, delta_p, outB0, outC, outDelta);
}

// ---------------- scan pass 1: per-chunk affine carry (s_out = P*s_in + Q) ----------------
__global__ void k_scan_pass1(const float* __restrict__ delta, const float* __restrict__ B0,
                             const float* __restrict__ x, const float* __restrict__ A,
                             float* __restrict__ P, float* __restrict__ Q) {
    int tid = blockIdx.x * blockDim.x + threadIdx.x;   // B*NCH*D = 131072
    int d = tid & (DMODEL - 1);
    int r = tid >> 10;
    int ch = r & (NCH - 1);
    int b = r >> 4;
    int l0 = ch * CHLEN;
    size_t base  = ((size_t)b * SEQL + l0) * DMODEL + d;
    size_t abase = (size_t)l0 * DMODEL + d;
    float p = 1.f, q = 0.f;
    for (int t = 0; t < CHLEN; ++t) {
        size_t idx = base + (size_t)t * DMODEL;
        float dl = delta[idx];
        float a  = dl * A[abase + (size_t)t * DMODEL];
        float bx = dl * B0[idx] * x[idx];
        q = fmaf(a, q, bx);
        p *= a;
    }
    P[tid] = p; Q[tid] = q;
}

// ---------------- scan pass 2: serial combine of NCH carries per (b,d) ----------------
__global__ void k_scan_pass2(const float* __restrict__ P, const float* __restrict__ Q,
                             float* __restrict__ Sin) {
    int tid = blockIdx.x * blockDim.x + threadIdx.x;   // 8192 = B*D
    int d = tid & (DMODEL - 1);
    int b = tid >> 10;
    float s = 0.f;
    for (int ch = 0; ch < NCH; ++ch) {
        int i = ((b * NCH + ch) << 10) + d;
        Sin[i] = s;
        s = fmaf(P[i], s, Q[i]);
    }
}

// ---------------- scan pass 3: replay with correct s_in, fuse T-gate and output gate ----------------
__global__ void k_scan_pass3(const float* __restrict__ delta, const float* __restrict__ B0,
                             const float* __restrict__ C, const float* __restrict__ x,
                             const float* __restrict__ A, const float* __restrict__ tf,
                             const float* __restrict__ Sin, float* __restrict__ out) {
    int tid = blockIdx.x * blockDim.x + threadIdx.x;   // 131072
    int d = tid & (DMODEL - 1);
    int r = tid >> 10;
    int ch = r & (NCH - 1);
    int b = r >> 4;
    int l0 = ch * CHLEN;
    size_t base  = ((size_t)b * SEQL + l0) * DMODEL + d;
    size_t abase = (size_t)l0 * DMODEL + d;
    float s = Sin[tid];
    float t_fac = tf[(b << 10) + d];
    for (int t = 0; t < CHLEN; ++t) {
        size_t idx = base + (size_t)t * DMODEL;
        float dl = delta[idx];
        float a  = dl * A[abase + (size_t)t * DMODEL];
        float xv = x[idx];
        float bx = dl * B0[idx] * xv;
        s = fmaf(a, s, bx);
        float y = fmaf(C[idx], s, t_fac * xv);
        out[idx] = y * sigmoidf_(xv);
    }
}

extern "C" void kernel_launch(void* const* d_in, const int* in_sizes, int n_in,
                              void* d_out, int out_size, void* d_ws, size_t ws_size,
                              hipStream_t stream) {
    const float* x       = (const float*)d_in[0];
    const float* W_B     = (const float*)d_in[1];
    const float* b_B     = (const float*)d_in[2];
    const float* W_C     = (const float*)d_in[3];
    const float* b_C     = (const float*)d_in[4];
    const float* W_d     = (const float*)d_in[5];
    const float* b_d     = (const float*)d_in[6];
    const float* delta_p = (const float*)d_in[7];
    const float* A       = (const float*)d_in[8];
    const float* W_st    = (const float*)d_in[9];
    const float* b_st    = (const float*)d_in[10];
    const float* st_bias = (const float*)d_in[11];
    float* out = (float*)d_out;

    const size_t NX = (size_t)MROWS * DMODEL;      // 16M elements
    const size_t NW = (size_t)DMODEL * DMODEL;     // 1M elements

    // workspace layout
    char* w = (char*)d_ws;
    __bf16* xb  = (__bf16*)w; w += NX * sizeof(__bf16);
    __bf16* wbB = (__bf16*)w; w += NW * sizeof(__bf16);
    __bf16* wbC = (__bf16*)w; w += NW * sizeof(__bf16);
    __bf16* wbD = (__bf16*)w; w += NW * sizeof(__bf16);
    float* dDelta = (float*)w; w += NX * sizeof(float);
    float* dB0    = (float*)w; w += NX * sizeof(float);
    float* dC     = (float*)w; w += NX * sizeof(float);
    float* sq  = (float*)w; w += (size_t)BATCH * DMODEL * sizeof(float);
    float* tf  = (float*)w; w += (size_t)BATCH * DMODEL * sizeof(float);
    float* P   = (float*)w; w += (size_t)BATCH * NCH * DMODEL * sizeof(float);
    float* Q   = (float*)w; w += (size_t)BATCH * NCH * DMODEL * sizeof(float);
    float* Sin = (float*)w; w += (size_t)BATCH * NCH * DMODEL * sizeof(float);

    // 1) fp32 -> bf16 for GEMM operands (x stays L2-resident at 32MB)
    k_f32_to_bf16<<<8192, 256, 0, stream>>>(x,   xb,  (int)NX);
    k_f32_to_bf16<<<1024, 256, 0, stream>>>(W_B, wbB, (int)NW);
    k_f32_to_bf16<<<1024, 256, 0, stream>>>(W_C, wbC, (int)NW);
    k_f32_to_bf16<<<1024, 256, 0, stream>>>(W_d, wbD, (int)NW);

    // 2) SToken path
    k_mean_seq<<<32, 256, 0, stream>>>(x, sq);
    k_stoken<<<32, 256, 0, stream>>>(sq, W_st, b_st, st_bias, tf);

    // 3) fused triple WMMA GEMM + delta sigmoid epilogue (2048 blocks x 8 waves)
    k_gemm3_wmma<<<2048, 256, 0, stream>>>(xb, wbB, wbC, wbD, b_B, b_C, b_d,
                                           delta_p, dB0, dC, dDelta);

    // 4) chunked parallel scan
    k_scan_pass1<<<512, 256, 0, stream>>>(dDelta, dB0, x, A, P, Q);
    k_scan_pass2<<<32, 256, 0, stream>>>(P, Q, Sin);
    k_scan_pass3<<<512, 256, 0, stream>>>(dDelta, dB0, dC, x, A, tf, Sin, out);
}
